// TransformerEncoder_52046413693399
// MI455X (gfx1250) — compile-verified, matrix operand order checked by reading, not run on hardware
//
#include <hip/hip_runtime.h>

constexpr int kE    = 512;
constexpr int kH    = 8;
constexpr int kHD   = 64;
constexpr int kS    = 2048;
constexpr int kN    = 2;
constexpr int kRows = kN * kS;
constexpr int kL    = 4;

typedef __attribute__((ext_vector_type(16))) __bf16 v16bf;
typedef __attribute__((ext_vector_type(8)))  float  v8f;

__device__ __forceinline__ v8f wmma_bf16(v16bf a, v16bf b, v8f c) {
  return __builtin_amdgcn_wmma_f32_16x16x32_bf16(false, a, false, b, (short)0, c,
                                                 false, false);
}

// Load one 16x32 bf16 A/B fragment. Lane l holds row/col (l&15); elements 0..7
// are K = ((l>>4)*8)+0..7, elements 8..15 are K = 16+((l>>4)*8)+0..7.
__device__ __forceinline__ v16bf ldfrag(const __bf16* p) {
  union { uint4 u[2]; v16bf v; } x;
  x.u[0] = *(const uint4*)p;
  x.u[1] = *(const uint4*)(p + 16);
  return x.v;
}

// Async global->LDS copy of 16 bytes (per-lane), tracked by ASYNCcnt.
__device__ __forceinline__ void async_copy_b128(void* lds_ptr, const void* gptr) {
  unsigned lds = (unsigned)(unsigned long long)lds_ptr;  // flat addr[31:0] == LDS offset
  asm volatile("global_load_async_to_lds_b128 %0, %1, off"
               :: "v"(lds), "v"(gptr) : "memory");
}
__device__ __forceinline__ void async_wait0() {
  asm volatile("s_wait_asynccnt 0x0" ::: "memory");
}

// ---------------------------------------------------------------------------
// C[4096,512] = A[4096,512] @ W[512,512]^T + bias
// AMODE: 0 = A is f32 (convert while staging), 1 = A is bf16 (async copy)
// OMODE: 0 = bf16 row-major, 1 = f32 row-major,
//        2 = bf16 transposed per-head: vt[n][h][d][s] (packed b128 stores)
// ---------------------------------------------------------------------------
template <int AMODE, int OMODE>
__global__ __launch_bounds__(256) void gemm_kernel(
    const void* __restrict__ Aptr, const float* __restrict__ W,
    const float* __restrict__ bias, __bf16* __restrict__ Cb,
    float* __restrict__ Cf) {
  __shared__ __bf16 As[128][32];
  __shared__ __bf16 Ws[128][32];
  const int tid  = threadIdx.x;
  const int wave = tid >> 5, lane = tid & 31;
  const int bm = blockIdx.x, bn = blockIdx.y;
  const int sr = tid >> 1;            // staging row 0..127
  const int sc = (tid & 1) << 4;      // staging col 0 or 16

  v8f acc[8];
  for (int i = 0; i < 8; ++i)
    for (int e = 0; e < 8; ++e) acc[i][e] = 0.f;

  for (int k0 = 0; k0 < kE; k0 += 32) {
    if (AMODE == 1) {
      const __bf16* src = (const __bf16*)Aptr + (size_t)(bm * 128 + sr) * kE + k0 + sc;
      async_copy_b128(&As[sr][sc], src);
      async_copy_b128(&As[sr][sc + 8], src + 8);
    } else {
      const float* src = (const float*)Aptr + (size_t)(bm * 128 + sr) * kE + k0 + sc;
#pragma unroll
      for (int j = 0; j < 16; j += 4) {
        float4 f = *(const float4*)(src + j);
        As[sr][sc + j + 0] = (__bf16)f.x; As[sr][sc + j + 1] = (__bf16)f.y;
        As[sr][sc + j + 2] = (__bf16)f.z; As[sr][sc + j + 3] = (__bf16)f.w;
      }
    }
    {
      const float* src = W + (size_t)(bn * 128 + sr) * kE + k0 + sc;
#pragma unroll
      for (int j = 0; j < 16; j += 4) {
        float4 f = *(const float4*)(src + j);
        Ws[sr][sc + j + 0] = (__bf16)f.x; Ws[sr][sc + j + 1] = (__bf16)f.y;
        Ws[sr][sc + j + 2] = (__bf16)f.z; Ws[sr][sc + j + 3] = (__bf16)f.w;
      }
    }
    if (AMODE == 1) async_wait0();
    __syncthreads();
    const int fr = lane & 15, fkb = (lane >> 4) << 3;
    v16bf af = ldfrag(&As[wave * 16 + fr][fkb]);
#pragma unroll
    for (int ns = 0; ns < 8; ++ns) {
      v16bf bf = ldfrag(&Ws[ns * 16 + fr][fkb]);
      acc[ns] = wmma_bf16(af, bf, acc[ns]);
    }
    __syncthreads();
  }

  // C/D layout: lanes 0-15 -> col=lane, rows e+0..7 ; lanes 16-31 -> rows 8+e
  const int mrow = bm * 128 + wave * 16 + ((lane >> 4) << 3);
  const int nc0  = bn * 128 + (lane & 15);
#pragma unroll
  for (int ns = 0; ns < 8; ++ns) {
    const int col = nc0 + ns * 16;
    const float bvl = bias[col];
    if (OMODE == 2) {
      // 8 accumulator elements = 8 consecutive s at fixed (h,d): one b128 store
      const int hh = col >> 6, dd = col & (kHD - 1);
      const int nn = mrow >> 11, ss = mrow & (kS - 1);
      union { uint4 u; __bf16 b[8]; } pk;
#pragma unroll
      for (int e = 0; e < 8; ++e) pk.b[e] = (__bf16)(acc[ns][e] + bvl);
      *(uint4*)(Cb + (((size_t)nn * kH + hh) * kHD + dd) * kS + ss) = pk.u;
    } else {
#pragma unroll
      for (int e = 0; e < 8; ++e) {
        float v = acc[ns][e] + bvl;
        size_t idx = (size_t)(mrow + e) * kE + col;
        if (OMODE == 1) Cf[idx] = v; else Cb[idx] = (__bf16)v;
      }
    }
  }
}

// ---------------------------------------------------------------------------
// Flash attention per (n, h, 128-query tile); 8 waves x 16 queries each.
// All staging (K, V-transposed, mask) is async global->LDS, double-buffered.
// ---------------------------------------------------------------------------
__global__ __launch_bounds__(256) void attn_kernel(
    const __bf16* __restrict__ qs, const __bf16* __restrict__ ks,
    const __bf16* __restrict__ vt,   // [n][h][d][s] bf16
    const int* __restrict__ mask, __bf16* __restrict__ outp) {
  __shared__ __bf16 Ksh[2][32][kHD];   // K block: [key][d]
  __shared__ __bf16 Vts[2][kHD][32];   // V block: [d][key]
  __shared__ int    Msk[2][128][32];   // mask tile: [q_local][key_local]
  __shared__ __bf16 Ps[8][16][32];     // per-wave P staging: [wave][q][key]
  const int tid  = threadIdx.x;
  const int wave = tid >> 5, lane = tid & 31;
  const int n = blockIdx.z, h = blockIdx.y, qt = blockIdx.x;
  const int fr  = lane & 15;
  const int fkb = (lane >> 4) << 3;   // K-element base within fragment
  const int m8  = (lane >> 4) << 3;   // row base within C layout
  const int q0  = qt * 128 + wave * 16;
  const size_t rowbase = (size_t)n * kS;
  const size_t vhbase  = ((size_t)n * kH + h) * kHD * kS;
  const size_t mbase   = ((size_t)n * kS + (size_t)(qt * 128)) * kS;

  auto stage = [&](int kb0, int buf) {
    const int key = tid >> 3, dseg = (tid & 7) << 3;
    async_copy_b128(&Ksh[buf][key][dseg],
                    ks + (rowbase + kb0 + key) * kE + h * kHD + dseg);
    const int dd = tid >> 2, seg = (tid & 3) << 3;
    async_copy_b128(&Vts[buf][dd][seg], vt + vhbase + (size_t)dd * kS + kb0 + seg);
    const int mr = tid >> 1, mc = (tid & 1) << 4;
    const int* mg = mask + mbase + (size_t)mr * kS + kb0 + mc;
    async_copy_b128(&Msk[buf][mr][mc + 0],  mg + 0);
    async_copy_b128(&Msk[buf][mr][mc + 4],  mg + 4);
    async_copy_b128(&Msk[buf][mr][mc + 8],  mg + 8);
    async_copy_b128(&Msk[buf][mr][mc + 12], mg + 12);
  };

  v16bf qf[2];
#pragma unroll
  for (int kc = 0; kc < 2; ++kc)
    qf[kc] = ldfrag(qs + (rowbase + q0 + fr) * kE + h * kHD + kc * 32 + fkb);

  v8f o[4];
  for (int d = 0; d < 4; ++d)
    for (int e = 0; e < 8; ++e) o[d][e] = 0.f;
  float mrow[8], lrow[8];
  for (int e = 0; e < 8; ++e) { mrow[e] = -3.0e38f; lrow[e] = 0.f; }
  const float scale = 0.04419417382415922f;  // 1/sqrt(512)

  stage(0, 0);
  async_wait0();
  __syncthreads();

  constexpr int kNB = kS / 32;
  for (int it = 0; it < kNB; ++it) {
    const int buf = it & 1;
    if (it + 1 < kNB) stage((it + 1) * 32, buf ^ 1);  // prefetch next block

    // scores: S[16q x 32keys] = Q(16x64) @ K^T, two 16x16 output tiles
    v8f s0, s1;
    for (int e = 0; e < 8; ++e) { s0[e] = 0.f; s1[e] = 0.f; }
#pragma unroll
    for (int kc = 0; kc < 2; ++kc) {
      v16bf k0f = ldfrag(&Ksh[buf][fr][kc * 32 + fkb]);
      s0 = wmma_bf16(qf[kc], k0f, s0);
      v16bf k1f = ldfrag(&Ksh[buf][16 + fr][kc * 32 + fkb]);
      s1 = wmma_bf16(qf[kc], k1f, s1);
    }

    // mask (reference: masked energy = -1e-10, then /sqrt(E)) + online softmax
    float sc0[8], sc1[8];
    {
      const int lr0 = wave * 16 + m8;
#pragma unroll
      for (int e = 0; e < 8; ++e) {
        float e0 = s0[e], e1 = s1[e];
        if (Msk[buf][lr0 + e][fr] == 1)      e0 = -1e-10f;
        if (Msk[buf][lr0 + e][16 + fr] == 1) e1 = -1e-10f;
        sc0[e] = e0 * scale; sc1[e] = e1 * scale;
      }
    }
    float alpha[8];
#pragma unroll
    for (int e = 0; e < 8; ++e) {
      float rm = fmaxf(sc0[e], sc1[e]);
      for (int off = 1; off < 16; off <<= 1)
        rm = fmaxf(rm, __shfl_xor(rm, off, 32));      // within 16-lane half
      float mn = fmaxf(mrow[e], rm);
      alpha[e] = __expf(mrow[e] - mn);
      mrow[e] = mn;
      float p0 = __expf(sc0[e] - mn), p1 = __expf(sc1[e] - mn);
      Ps[wave][m8 + e][fr]      = (__bf16)p0;
      Ps[wave][m8 + e][16 + fr] = (__bf16)p1;
      float rs = p0 + p1;
      for (int off = 1; off < 16; off <<= 1)
        rs += __shfl_xor(rs, off, 32);
      lrow[e] = lrow[e] * alpha[e] + rs;
    }
    // P was written cross-lane within this wave; order DS store -> DS load
    asm volatile("s_wait_dscnt 0" ::: "memory");
    v16bf pf = ldfrag(&Ps[wave][fr][fkb]);
#pragma unroll
    for (int d = 0; d < 4; ++d) {
      for (int e = 0; e < 8; ++e) o[d][e] *= alpha[e];
      v16bf vf = ldfrag(&Vts[buf][d * 16 + fr][fkb]);
      o[d] = wmma_bf16(pf, vf, o[d]);
    }
    async_wait0();    // this wave's prefetch for buf^1 has landed
    __syncthreads();  // all waves done with buf & all prefetches visible
  }

#pragma unroll
  for (int d = 0; d < 4; ++d)
#pragma unroll
    for (int e = 0; e < 8; ++e) {
      float v = o[d][e] / lrow[e];
      outp[(rowbase + q0 + m8 + e) * kE + h * kHD + d * 16 + fr] = (__bf16)v;
    }
}

// ---------------------------------------------------------------------------
// Fused residual + LayerNorm: one block per row of 512
// ---------------------------------------------------------------------------
__global__ __launch_bounds__(256) void ln_kernel(
    const float* __restrict__ proj, const float* __restrict__ xin,
    const float* __restrict__ g, const float* __restrict__ b,
    float* __restrict__ xout) {
  __shared__ float red[8];
  const int row = blockIdx.x, tid = threadIdx.x;
  const int wave = tid >> 5, lane = tid & 31;
  const size_t base = (size_t)row * kE;
  float x0 = proj[base + tid] + xin[base + tid];
  float x1 = proj[base + tid + 256] + xin[base + tid + 256];
  float s = x0 + x1;
  for (int off = 16; off > 0; off >>= 1) s += __shfl_xor(s, off, 32);
  if (lane == 0) red[wave] = s;
  __syncthreads();
  float tot = 0.f;
  for (int i = 0; i < 8; ++i) tot += red[i];
  const float mu = tot * (1.f / kE);
  __syncthreads();
  const float d0 = x0 - mu, d1 = x1 - mu;
  float v = d0 * d0 + d1 * d1;
  for (int off = 16; off > 0; off >>= 1) v += __shfl_xor(v, off, 32);
  if (lane == 0) red[wave] = v;
  __syncthreads();
  float vt = 0.f;
  for (int i = 0; i < 8; ++i) vt += red[i];
  const float rstd = rsqrtf(vt * (1.f / kE) + 1e-5f);
  xout[base + tid]       = d0 * rstd * g[tid]       + b[tid];
  xout[base + tid + 256] = d1 * rstd * g[tid + 256] + b[tid + 256];
}

// ---------------------------------------------------------------------------
extern "C" void kernel_launch(void* const* d_in, const int* in_sizes, int n_in,
                              void* d_out, int out_size, void* d_ws,
                              size_t ws_size, hipStream_t stream) {
  (void)in_sizes; (void)n_in; (void)out_size; (void)ws_size;
  const float* query = (const float*)d_in[0];
  const float* value = (const float*)d_in[1];
  const int*   mask  = (const int*)d_in[2];
  const float* Wq = (const float*)d_in[3];
  const float* bq = (const float*)d_in[4];
  const float* Wk = (const float*)d_in[5];
  const float* bk = (const float*)d_in[6];
  const float* Wv = (const float*)d_in[7];
  const float* bv = (const float*)d_in[8];
  const float* Wo = (const float*)d_in[9];
  const float* bo = (const float*)d_in[10];
  const float* lg = (const float*)d_in[11];
  const float* lb = (const float*)d_in[12];

  char* w = (char*)d_ws;
  float*  X    = (float*)w;  w += (size_t)kRows * kE * sizeof(float);   // 8 MB
  __bf16* qsb  = (__bf16*)w; w += (size_t)kRows * kE * sizeof(__bf16);  // 4 MB
  __bf16* ksb  = (__bf16*)w; w += (size_t)kRows * kE * sizeof(__bf16);  // 4 MB
  __bf16* vtb  = (__bf16*)w; w += (size_t)kRows * kE * sizeof(__bf16);  // 4 MB (transposed per-head)
  __bf16* atb  = (__bf16*)w; w += (size_t)kRows * kE * sizeof(__bf16);  // 4 MB
  float*  proj = (float*)w;                                             // 8 MB

  const dim3 gg(kRows / 128, kE / 128), gb(256);
  const dim3 ga(kS / 128, kH, kN);

  for (int i = 0; i < kL; ++i) {
    const float* xin = (i == 0) ? query : X;
    const size_t wo = (size_t)i * kE * kE;
    gemm_kernel<0, 0><<<gg, gb, 0, stream>>>(xin,   Wq + wo, bq + i * kE, qsb, nullptr);
    gemm_kernel<0, 0><<<gg, gb, 0, stream>>>(value, Wk + wo, bk + i * kE, ksb, nullptr);
    gemm_kernel<0, 2><<<gg, gb, 0, stream>>>(value, Wv + wo, bv + i * kE, vtb, nullptr);
    attn_kernel<<<ga, gb, 0, stream>>>(qsb, ksb, vtb, mask, atb);
    gemm_kernel<1, 1><<<gg, gb, 0, stream>>>(atb, Wo + wo, bo + i * kE, nullptr, proj);
    float* xout = (i == kL - 1) ? (float*)d_out : X;
    ln_kernel<<<kRows, 256, 0, stream>>>(proj, xin, lg + i * kE, lb + i * kE, xout);
  }
}
